// LSTM_39333310496826
// MI455X (gfx1250) — compile-verified, matrix operand order checked by reading, not run on hardware
//
#include <hip/hip_runtime.h>
#include <math.h>

typedef __bf16 bf16_t;
typedef __attribute__((ext_vector_type(16))) __bf16 v16bf;
typedef __attribute__((ext_vector_type(8)))  __bf16 bf8;
typedef __attribute__((ext_vector_type(8)))  float  v8f;

#define HDIM 1024
#define GDIM 4096   // 4*H
#define TSEQ 2048
#define NCLS 512
#define RBLK 32     // persistent blocks in recurrent kernel
#define RTHR 256    // 8 waves -> 8 tiles of 16 cols = 128 cols per WG
#define LPAD 8      // bf16 elements of padding per LDS column (16B -> 4-bank lane spread)
#define LSTRIDE (HDIM + LPAD)                    // 1032 elements per column
#define RLDS_BYTES (128 * LSTRIDE * 2)           // 264192 B < 320KB WGP LDS

// ---- fp32 -> bf16 round-to-nearest-even ----
__device__ __forceinline__ bf16_t f2bf(float f) {
  unsigned u = __float_as_uint(f);
  u += 0x7FFFu + ((u >> 16) & 1u);
  unsigned short h = (unsigned short)(u >> 16);
  return __builtin_bit_cast(bf16_t, h);
}

// Assemble a 16-bf16 WMMA fragment from two 16B global loads (ISA 16-bit A/B layout:
// elems 0-7 = K base, elems 8-15 = K base + 16; lanes 16-31 shifted by kgrp=8).
__device__ __forceinline__ v16bf ldfrag_g(const bf16_t* __restrict__ p) {
  bf8 lo = *(const bf8*)p;
  bf8 hi = *(const bf8*)(p + 16);
  v16bf r;
#pragma unroll
  for (int e = 0; e < 8; e++) { r[e] = lo[e]; r[e + 8] = hi[e]; }
  return r;
}

__global__ void cvt_bf16_kernel(const float* __restrict__ s, bf16_t* __restrict__ d, int n) {
  int i = blockIdx.x * blockDim.x + threadIdx.x;
  if (i < n) d[i] = f2bf(s[i]);
}

__global__ void combine_bias_kernel(const float* __restrict__ a, const float* __restrict__ b,
                                    float* __restrict__ o, int n) {
  int i = blockIdx.x * blockDim.x + threadIdx.x;
  if (i < n) o[i] = a[i] + b[i];
}

__global__ void init_layer_kernel(float* __restrict__ c, bf16_t* __restrict__ hzero,
                                  unsigned* __restrict__ bar) {
  int i = blockIdx.x * blockDim.x + threadIdx.x;
  if (i < HDIM) { c[i] = 0.f; hzero[i] = f2bf(0.f); }
  if (i == 0) *bar = 0u;
}

// ---- bf16 WMMA GEMM, register-blocked: each wave computes a 32x64 tile ----
// C[M,N] = A[M,K] @ Bw[N,K]^T + bias[N].  2 A-frags x 4 B-frags -> 8 WMMAs / 12 loads.
__global__ void wmma_gemm_bias_kernel(const bf16_t* __restrict__ A, const bf16_t* __restrict__ Bw,
                                      const float* __restrict__ bias, float* __restrict__ C,
                                      int M, int N, int K) {
  int wave = (int)((blockIdx.x * blockDim.x + threadIdx.x) >> 5);
  int lane = threadIdx.x & 31;
  int nsup = N >> 6;                      // 64-col supertiles
  int sm = wave / nsup;
  int sn = wave - sm * nsup;
  if (sm * 32 >= M) return;               // uniform per wave: no partial-EXEC WMMA

  int l15  = lane & 15;
  int kgrp = (lane >> 4) << 3;            // 0 (lanes 0-15) or 8 (lanes 16-31)
  const bf16_t* arow0 = A + (size_t)(sm * 32 + l15) * K;
  const bf16_t* arow1 = arow0 + (size_t)16 * K;
  const bf16_t* brow[4];
#pragma unroll
  for (int j = 0; j < 4; j++) brow[j] = Bw + (size_t)(sn * 64 + j * 16 + l15) * K;

  v8f acc[2][4] = {};
  for (int kk = 0; kk < K; kk += 32) {
    int k0 = kk + kgrp;
    __builtin_prefetch(arow0 + k0 + 128, 0, 1);   // global_prefetch_b8
    __builtin_prefetch(arow1 + k0 + 128, 0, 1);
    v16bf a0 = ldfrag_g(arow0 + k0);
    v16bf a1 = ldfrag_g(arow1 + k0);
#pragma unroll
    for (int j = 0; j < 4; j++) {
      __builtin_prefetch(brow[j] + k0 + 128, 0, 1);
      v16bf bv = ldfrag_g(brow[j] + k0);
      acc[0][j] = __builtin_amdgcn_wmma_f32_16x16x32_bf16(false, a0, false, bv, (short)0, acc[0][j], false, false);
      acc[1][j] = __builtin_amdgcn_wmma_f32_16x16x32_bf16(false, a1, false, bv, (short)0, acc[1][j], false, false);
    }
  }

  int hi8 = (lane >> 4) << 3;
#pragma unroll
  for (int i = 0; i < 2; i++) {
    int mbase = sm * 32 + i * 16 + hi8;
#pragma unroll
    for (int j = 0; j < 4; j++) {
      int col = sn * 64 + j * 16 + l15;
      float bc = bias ? bias[col] : 0.f;
#pragma unroll
      for (int r = 0; r < 8; r++)
        C[(size_t)(mbase + r) * N + col] = acc[i][j][r] + bc;
    }
  }
}

// ---- grid-wide barrier (persistent kernel, monotonically increasing phase) ----
__device__ __forceinline__ void grid_barrier(unsigned* bar, unsigned& phase) {
  __threadfence();
  __syncthreads();
  if (threadIdx.x == 0) {
    phase += 1u;
    __hip_atomic_fetch_add(bar, 1u, __ATOMIC_ACQ_REL, __HIP_MEMORY_SCOPE_AGENT);
    while (__hip_atomic_load(bar, __ATOMIC_ACQUIRE, __HIP_MEMORY_SCOPE_AGENT) < phase * RBLK)
      __builtin_amdgcn_s_sleep(1);
  }
  __syncthreads();
  __threadfence();
}

// ---- persistent recurrent kernel ----
// 32 WGs x 256 thr (8 waves). WG b owns gate cols [b*128, b*128+128). Its W_hh slice
// (128 x 1024 bf16, +8-elem column pad) lives in 258KB of LDS for the whole sequence.
// Per step: wave w computes 16 gate cols via WMMA (h broadcast into all 16 A-rows,
// B-fragments from LDS), 4 independent accumulator chains; grid barrier; 32 hidden
// units per WG run the LSTM cell; grid barrier.
__global__ void lstm_recur_kernel(const float* __restrict__ gx, const bf16_t* __restrict__ Whh,
                                  float* __restrict__ gates, float* __restrict__ cst,
                                  const bf16_t* __restrict__ h0, bf16_t* __restrict__ hseq,
                                  unsigned* __restrict__ bar, int T) {
  extern __shared__ bf16_t lds[];          // 128 * LSTRIDE bf16
  int tid  = threadIdx.x;
  int blk  = blockIdx.x;
  int wIn  = tid >> 5;                     // wave in block, 0..7
  int lane = tid & 31;
  int l15  = lane & 15;
  int kgrp = (lane >> 4) << 3;

  // One-time LDS fill: 128 cols x 1024 elems in 16B chunks (ds_store_b128).
  for (int idx = tid; idx < 128 * (HDIM / 8); idx += RTHR) {
    int c = idx >> 7;                      // 0..127  (HDIM/8 = 128 segments)
    int s = idx & 127;
    bf8 v = *(const bf8*)(Whh + (size_t)(blk * 128 + c) * HDIM + s * 8);
    *(bf8*)(&lds[c * LSTRIDE + s * 8]) = v;
  }
  __syncthreads();

  int colLocal = wIn * 16 + l15;           // this lane's gate column within WG
  int col      = blk * 128 + colLocal;
  unsigned phase = 0;

  for (int t = 0; t < T; t++) {
    const bf16_t* hprev = (t == 0) ? h0 : (hseq + (size_t)(t - 1) * HDIM);
    const float*  gxt   = gx + (size_t)t * GDIM;

    // ---- phase A: gate pre-activation via WMMA (B from LDS) ----
    v8f acc[4] = {};
#pragma unroll
    for (int kk = 0; kk < HDIM; kk += 32) {
      int k0 = kk + kgrp;
      v16bf av = ldfrag_g(hprev + k0);     // broadcast h into all 16 A-rows
      bf8 blo = *(const bf8*)(&lds[colLocal * LSTRIDE + k0]);
      bf8 bhi = *(const bf8*)(&lds[colLocal * LSTRIDE + k0 + 16]);
      v16bf bv;
#pragma unroll
      for (int e = 0; e < 8; e++) { bv[e] = blo[e]; bv[e + 8] = bhi[e]; }
      int ch = (kk >> 5) & 3;              // 4 independent accumulation chains
      acc[ch] = __builtin_amdgcn_wmma_f32_16x16x32_bf16(false, av, false, bv, (short)0, acc[ch], false, false);
    }
    if (lane < 16)                         // row M=0 lives in VGPR0, lanes 0-15
      gates[col] = acc[0][0] + acc[1][0] + acc[2][0] + acc[3][0] + gxt[col];

    grid_barrier(bar, phase);

    // ---- phase B: LSTM cell update, 32 hidden units per WG ----
    if (tid < 32) {
      int j = blk * 32 + tid;
      float gi = gates[j];
      float gf = gates[j + HDIM];
      float gg = gates[j + 2 * HDIM];
      float go = gates[j + 3 * HDIM];
      gi = 1.f / (1.f + __expf(-gi));
      gf = 1.f / (1.f + __expf(-gf));
      gg = tanhf(gg);
      go = 1.f / (1.f + __expf(-go));
      float cv = gf * cst[j] + gi * gg;
      cst[j] = cv;
      hseq[(size_t)t * HDIM + j] = f2bf(go * tanhf(cv));
    }

    grid_barrier(bar, phase);
  }
}

extern "C" void kernel_launch(void* const* d_in, const int* in_sizes, int n_in,
                              void* d_out, int out_size, void* d_ws, size_t ws_size,
                              hipStream_t stream) {
  (void)in_sizes; (void)n_in; (void)out_size; (void)ws_size;
  const float* x    = (const float*)d_in[0];
  const float* Wih[3] = {(const float*)d_in[1], (const float*)d_in[5], (const float*)d_in[9]};
  const float* Whh[3] = {(const float*)d_in[2], (const float*)d_in[6], (const float*)d_in[10]};
  const float* bih[3] = {(const float*)d_in[3], (const float*)d_in[7], (const float*)d_in[11]};
  const float* bhh[3] = {(const float*)d_in[4], (const float*)d_in[8], (const float*)d_in[12]};
  const float* fcw = (const float*)d_in[13];
  const float* fcb = (const float*)d_in[14];
  float* out = (float*)d_out;

  // ---- workspace layout (~64 MB) ----
  char* p = (char*)d_ws;
  auto take = [&](size_t bytes) { char* r = p; p += (bytes + 255) & ~(size_t)255; return r; };
  bf16_t*  bf_x    = (bf16_t*)take((size_t)TSEQ * HDIM * 2);
  bf16_t*  bf_hA   = (bf16_t*)take((size_t)TSEQ * HDIM * 2);
  bf16_t*  bf_hB   = (bf16_t*)take((size_t)TSEQ * HDIM * 2);
  bf16_t*  bf_Wih  = (bf16_t*)take((size_t)GDIM * HDIM * 2);
  bf16_t*  bf_Whh  = (bf16_t*)take((size_t)GDIM * HDIM * 2);
  bf16_t*  bf_fcw  = (bf16_t*)take((size_t)NCLS * HDIM * 2);
  float*   gx      = (float*) take((size_t)TSEQ * GDIM * 4);
  float*   biasbuf = (float*) take((size_t)GDIM * 4);
  float*   gatesws = (float*) take((size_t)GDIM * 4);
  float*   cstate  = (float*) take((size_t)HDIM * 4);
  bf16_t*  hzero   = (bf16_t*)take((size_t)HDIM * 2);
  unsigned* bar    = (unsigned*)take(256);

  // Allow >64KB dynamic LDS for the persistent recurrent kernel (WGP has 320KB).
  (void)hipFuncSetAttribute((const void*)lstm_recur_kernel,
                            hipFuncAttributeMaxDynamicSharedMemorySize, RLDS_BYTES);

  const int CT = 256;
  auto cvt = [&](const float* s, bf16_t* d, int n) {
    cvt_bf16_kernel<<<(n + CT - 1) / CT, CT, 0, stream>>>(s, d, n);
  };

  cvt(x,   bf_x,   TSEQ * HDIM);
  cvt(fcw, bf_fcw, NCLS * HDIM);

  bf16_t* hcur = bf_x;
  for (int l = 0; l < 3; l++) {
    cvt(Wih[l], bf_Wih, GDIM * HDIM);
    cvt(Whh[l], bf_Whh, GDIM * HDIM);
    combine_bias_kernel<<<(GDIM + CT - 1) / CT, CT, 0, stream>>>(bih[l], bhh[l], biasbuf, GDIM);

    // gx[T,4H] = hcur[T,H] @ Wih^T + (b_ih + b_hh); 32x64 tile per wave
    {
      int waves  = (TSEQ / 32) * (GDIM / 64);
      int blocks = (waves + 7) / 8;
      wmma_gemm_bias_kernel<<<blocks, 256, 0, stream>>>(hcur, bf_Wih, biasbuf, gx,
                                                        TSEQ, GDIM, HDIM);
    }

    bf16_t* hnext = (l & 1) ? bf_hB : bf_hA;
    init_layer_kernel<<<(HDIM + CT - 1) / CT, CT, 0, stream>>>(cstate, hzero, bar);
    lstm_recur_kernel<<<RBLK, RTHR, RLDS_BYTES, stream>>>(gx, bf_Whh, gatesws, cstate,
                                                          hzero, hnext, bar, TSEQ);
    hcur = hnext;
  }

  // out[T,NC] = h3[T,H] @ fc_w^T + fc_b
  {
    int waves  = (TSEQ / 32) * (NCLS / 64);
    int blocks = (waves + 7) / 8;
    wmma_gemm_bias_kernel<<<blocks, 256, 0, stream>>>(hcur, bf_fcw, fcb, out,
                                                      TSEQ, NCLS, HDIM);
  }
}
